// HRMMLA_34935263986345
// MI455X (gfx1250) — matrix operand order repaired and checked by reading, not compiled, not run on hardware
//
#include <hip/hip_runtime.h>
#include <math.h>

// ---------------------------------------------------------------- constants
#define S_LEN   2048
#define H_DIM   2048
#define N_HEADS 16
#define QLORA   1536
#define KVLORA  512
#define DNOPE   128
#define DROPE   64
#define DV      128
#define DQK     192     // DNOPE + DROPE
#define EPSV    1e-6f

typedef __attribute__((ext_vector_type(16))) __bf16 v16bf;
typedef __attribute__((ext_vector_type(8)))  __bf16 v8bf;
typedef __attribute__((ext_vector_type(8)))  float  v8f;

// ------------------------------------------------- async global->LDS (CDNA5)
// GLOBAL_LOAD_ASYNC_TO_LDS_B128: per-lane 16B copy memory->LDS, tracked by
// ASYNCcnt (CDNA5 ISA op 98).
// Probe round 2 showed the builtin exists with params:
//   (int4_vs addrspace(1)*, <param1>, Ii, Ii)   [addrspace(1) prints as __device__]
typedef int v4i_vs __attribute__((vector_size(16)));

#if defined(__AMDGCN__)
  #if __has_builtin(__builtin_amdgcn_global_load_async_to_lds_b128)
    #define ASYNC_B128(ldsp, gp)                                                    \
      __builtin_amdgcn_global_load_async_to_lds_b128(                               \
          (__attribute__((address_space(1))) v4i_vs*)(uintptr_t)(gp),               \
          (__attribute__((address_space(3))) v4i_vs*)(unsigned)(uintptr_t)(ldsp),   \
          0, 0)
  #else
    #define ASYNC_B128(ldsp, gp)                                                    \
      asm volatile("global_load_async_to_lds_b128 %0, %1, off"                      \
                   :: "v"((unsigned)(uintptr_t)(ldsp)), "v"(gp) : "memory")
  #endif
  #if __has_builtin(__builtin_amdgcn_s_wait_asynccnt)
    #define ASYNC_WAIT() __builtin_amdgcn_s_wait_asynccnt(0)
  #else
    #define ASYNC_WAIT() asm volatile("s_wait_asynccnt 0x0" ::: "memory")
  #endif
#else  // host pass: bodies only need to parse
  #define ASYNC_B128(ldsp, gp) (*(uint4*)(ldsp) = *(const uint4*)(gp))
  #define ASYNC_WAIT() ((void)0)
#endif

__device__ __forceinline__ v8f v8f_zero() {
  v8f z;
#pragma unroll
  for (int i = 0; i < 8; ++i) z[i] = 0.0f;
  return z;
}

__device__ __forceinline__ v16bf make_frag(v8bf lo, v8bf hi) {
  v16bf r;
#pragma unroll
  for (int i = 0; i < 8; ++i) { r[i] = lo[i]; r[i + 8] = hi[i]; }
  return r;
}

// A-operand fragment (16 rows x 32 K): lane L holds row (L&15); element halves
// are K {kb..kb+7} and {kb+16..kb+23} with kb=(L>=16)*8  [CDNA5 ISA 7.12.2]
__device__ __forceinline__ v16bf load_frag_a(const __bf16* base, int row, int stride) {
  const int lane = threadIdx.x & 31;
  const __bf16* p = base + (size_t)(row + (lane & 15)) * stride + ((lane >> 4) << 3);
  return make_frag(*(const v8bf*)p, *(const v8bf*)(p + 16));
}

// B-operand fragment (32 K x 16 cols), storage is (N,K) row-major: lane L holds
// col (L&15); K {kb..kb+7} and {kb+16..kb+23} with kb=(L>=16)*8
__device__ __forceinline__ v16bf load_frag_b(const __bf16* base, int row, int stride) {
  const int lane = threadIdx.x & 31;
  const __bf16* p = base + (size_t)(row + (lane & 15)) * stride + ((lane >> 4) << 3);
  return make_frag(*(const v8bf*)p, *(const v8bf*)(p + 16));
}

// ---------------------------------------------------------------- f32 -> bf16
__global__ __launch_bounds__(256) void cvt_f32_bf16(const float* __restrict__ in,
                                                    __bf16* __restrict__ out, int n) {
  for (int i = blockIdx.x * 256 + threadIdx.x; i < n; i += gridDim.x * 256)
    out[i] = (__bf16)in[i];
}

// ---------------------------------------------------------------- RMSNorm
__global__ __launch_bounds__(256) void rmsnorm_bf16(const float* __restrict__ in,
                                                    const float* __restrict__ w,
                                                    __bf16* __restrict__ out,
                                                    int cols, int in_stride) {
  const int row = blockIdx.x;
  const float* x = in + (size_t)row * in_stride;
  __shared__ float red[256];
  float acc = 0.0f;
  for (int c = threadIdx.x; c < cols; c += 256) { float v = x[c]; acc += v * v; }
  red[threadIdx.x] = acc;
  __syncthreads();
  for (int o = 128; o > 0; o >>= 1) {
    if ((int)threadIdx.x < o) red[threadIdx.x] += red[threadIdx.x + o];
    __syncthreads();
  }
  const float scale = rsqrtf(red[0] / (float)cols + EPSV);
  for (int c = threadIdx.x; c < cols; c += 256)
    out[(size_t)row * cols + c] = (__bf16)(x[c] * scale * w[c]);
}

// ---------------------------------------------------------------- GEMM (WMMA)
// C(MxN,f32) = A(MxK,bf16 row-major) * B(NxK,bf16 row-major)^T
// 256 thr = 8 waves, macro tile 64x128, wave tile 32x32, K-step 32.
// Double-buffered async global->LDS staging: loads of tile i+1 overlap WMMAs
// of tile i; one workgroup barrier per K-step.
__global__ __launch_bounds__(256) void gemm_bf16_nt(const __bf16* __restrict__ A,
                                                    const __bf16* __restrict__ B,
                                                    float* __restrict__ C,
                                                    int M, int N, int K) {
  __shared__ alignas(16) __bf16 As[2][64 * 32];
  __shared__ alignas(16) __bf16 Bs[2][128 * 32];
  const int tid  = threadIdx.x;
  const int lane = tid & 31;
  const int wave = tid >> 5;
  const int m0 = blockIdx.y * 64;
  const int n0 = blockIdx.x * 128;
  const int wm = (wave & 1) * 32;
  const int wn = (wave >> 1) * 32;

  v8f acc[2][2];
#pragma unroll
  for (int i = 0; i < 2; ++i)
#pragma unroll
    for (int j = 0; j < 2; ++j) acc[i][j] = v8f_zero();

  const int arow = tid >> 2, acol = (tid & 3) << 3;   // 64 rows x 32 : 8 elem/thr
  const int brow = tid >> 1, bcol = (tid & 1) << 4;   // 128 rows x 32: 16 elem/thr
  int bg = n0 + brow;
  if (bg >= N) bg = N - 1;  // clamp: junk lands only in cols >= N (never stored)

  const __bf16* agp = A + (size_t)(m0 + arow) * K + acol;
  const __bf16* bgp = B + (size_t)bg * K + bcol;

  auto issue = [&](int k0, int buf) {
    ASYNC_B128(&As[buf][arow * 32 + acol], agp + k0);
    ASYNC_B128(&Bs[buf][brow * 32 + bcol], bgp + k0);
    ASYNC_B128(&Bs[buf][brow * 32 + bcol + 8], bgp + k0 + 8);
  };

  issue(0, 0);
  int s = 0;
  for (int k0 = 0; k0 < K; k0 += 32, s ^= 1) {
    ASYNC_WAIT();      // my async copies for buffer s are in LDS
    __syncthreads();   // everyone's are
    if (k0 + 32 < K) {
      issue(k0 + 32, s ^ 1);
      if (k0 + 64 < K) {  // speculative prefetch two tiles ahead (near cache)
        __builtin_prefetch(agp + k0 + 64, 0, 3);
        __builtin_prefetch(bgp + k0 + 64, 0, 3);
      }
    }
    v16bf af0 = load_frag_a(As[s], wm + 0, 32);
    v16bf af1 = load_frag_a(As[s], wm + 16, 32);
    v16bf bf0 = load_frag_b(Bs[s], wn + 0, 32);
    v16bf bf1 = load_frag_b(Bs[s], wn + 16, 32);
    acc[0][0] = __builtin_amdgcn_wmma_f32_16x16x32_bf16(false, af0, false, bf0, (short)0, acc[0][0], false, false);
    acc[0][1] = __builtin_amdgcn_wmma_f32_16x16x32_bf16(false, af0, false, bf1, (short)0, acc[0][1], false, false);
    acc[1][0] = __builtin_amdgcn_wmma_f32_16x16x32_bf16(false, af1, false, bf0, (short)0, acc[1][0], false, false);
    acc[1][1] = __builtin_amdgcn_wmma_f32_16x16x32_bf16(false, af1, false, bf1, (short)0, acc[1][1], false, false);
  }

  const int mloc = (lane >> 4) * 8, nloc = lane & 15;
#pragma unroll
  for (int i = 0; i < 2; ++i)
#pragma unroll
    for (int j = 0; j < 2; ++j) {
      const int ng = n0 + wn + j * 16 + nloc;
      if (ng < N) {
#pragma unroll
        for (int v = 0; v < 8; ++v) {
          const int mg = m0 + wm + i * 16 + v + mloc;
          C[(size_t)mg * N + ng] = acc[i][j][v];
        }
      }
    }
}

// ---------------------------------------------------------------- RoPE + pack
__global__ __launch_bounds__(256) void rope_pack(const float* __restrict__ q,      // (S, 16*192)
                                                 const float* __restrict__ kv,     // (S, 576)
                                                 const float* __restrict__ kvp,    // (S, 16*256)
                                                 const float* __restrict__ cosb,   // (S, 64)
                                                 const float* __restrict__ sinb,   // (S, 64)
                                                 __bf16* __restrict__ Qh,
                                                 __bf16* __restrict__ Kh,
                                                 __bf16* __restrict__ Vt) {
  const int s = blockIdx.x;
  const float* cs = cosb + (size_t)s * DROPE;
  const float* sn = sinb + (size_t)s * DROPE;
  for (int idx = threadIdx.x; idx < N_HEADS * DQK; idx += 256) {
    const int h = idx / DQK, d = idx % DQK;
    float qv, kvv;
    if (d < DNOPE) {
      qv  = q[(size_t)s * (N_HEADS * DQK) + h * DQK + d];
      kvv = kvp[(size_t)s * (N_HEADS * 256) + h * 256 + d];
    } else {
      const int j = d - DNOPE;
      const float qa = q[(size_t)s * (N_HEADS * DQK) + h * DQK + DNOPE + j];
      const float qb = (j < DROPE / 2)
                           ? -q[(size_t)s * (N_HEADS * DQK) + h * DQK + DNOPE + j + DROPE / 2]
                           :  q[(size_t)s * (N_HEADS * DQK) + h * DQK + DNOPE + j - DROPE / 2];
      qv = qa * cs[j] + qb * sn[j];
      const float ka = kv[(size_t)s * 576 + KVLORA + j];
      const float kb = (j < DROPE / 2) ? -kv[(size_t)s * 576 + KVLORA + j + DROPE / 2]
                                       :  kv[(size_t)s * 576 + KVLORA + j - DROPE / 2];
      kvv = ka * cs[j] + kb * sn[j];
    }
    Qh[(size_t)h * S_LEN * DQK + (size_t)s * DQK + d] = (__bf16)qv;
    Kh[(size_t)h * S_LEN * DQK + (size_t)s * DQK + d] = (__bf16)kvv;
  }
  for (int idx = threadIdx.x; idx < N_HEADS * DV; idx += 256) {
    const int h = idx / DV, d = idx % DV;
    Vt[(size_t)h * DV * S_LEN + (size_t)d * S_LEN + s] =
        (__bf16)kvp[(size_t)s * (N_HEADS * 256) + h * 256 + DNOPE + d];
  }
}

// ---------------------------------------------------------------- flash attn
// grid(x=S/64, y=NH), 128 threads = 4 waves; wave owns 16 query rows.
// Double-buffered async K/V tiles; per 32-key block: 12 score WMMAs + 8 P*V.
__global__ __launch_bounds__(128) void flash_attn(const __bf16* __restrict__ Qh,
                                                  const __bf16* __restrict__ Kh,
                                                  const __bf16* __restrict__ Vt,
                                                  const float* __restrict__ mask, // (S,S)
                                                  __bf16* __restrict__ X,         // (S, 16*128)
                                                  float scale) {
  __shared__ alignas(16) __bf16 Ks[2][32 * DQK];   // 32 keys x 192
  __shared__ alignas(16) __bf16 Vs[2][DV * 32];    // 128 dims x 32 keys (t-major)
  __shared__ alignas(16) __bf16 Ps[4][16 * 32];    // per-wave P tile
  const int h    = blockIdx.y;
  const int tid  = threadIdx.x;
  const int wave = tid >> 5;
  const int lane = tid & 31;
  const int q0   = blockIdx.x * 64 + wave * 16;
  const int mloc = (lane >> 4) * 8, nloc = lane & 15;

  const __bf16* Qb = Qh + (size_t)h * S_LEN * DQK;
  const __bf16* Kb = Kh + (size_t)h * S_LEN * DQK;
  const __bf16* Vb = Vt + (size_t)h * DV * S_LEN + (size_t)tid * S_LEN;

  v16bf qf[6];
#pragma unroll
  for (int kb = 0; kb < 6; ++kb) qf[kb] = load_frag_a(Qb + kb * 32, q0, DQK);

  v8f o[8];
#pragma unroll
  for (int j = 0; j < 8; ++j) o[j] = v8f_zero();
  float m_run[8], l_run[8];
#pragma unroll
  for (int v = 0; v < 8; ++v) { m_run[v] = -3.0e38f; l_run[v] = 0.0f; }

  auto issue = [&](int t0, int buf) {
    // K tile: 32 contiguous rows of 192 bf16 -> flat 768-chunk copy
    const uint4* ksrc = (const uint4*)(Kb + (size_t)t0 * DQK);
    uint4* kdst = (uint4*)Ks[buf];
#pragma unroll
    for (int i = 0; i < 6; ++i) ASYNC_B128(kdst + tid + i * 128, ksrc + tid + i * 128);
    // V tile: one d-row (32 keys) per thread
    const uint4* vsrc = (const uint4*)(Vb + t0);
    uint4* vdst = (uint4*)(Vs[buf] + tid * 32);
#pragma unroll
    for (int c = 0; c < 4; ++c) ASYNC_B128(vdst + c, vsrc + c);
  };

  issue(0, 0);
  int s = 0;
  for (int it = 0; it < S_LEN / 32; ++it, s ^= 1) {
    const int t0 = it * 32;
    ASYNC_WAIT();
    __syncthreads();
    if (it + 1 < S_LEN / 32) issue(t0 + 32, s ^ 1);

    v8f s0 = v8f_zero(), s1 = v8f_zero();
#pragma unroll
    for (int kb = 0; kb < 6; ++kb) {
      v16bf b0 = load_frag_b(Ks[s] + kb * 32, 0, DQK);
      v16bf b1 = load_frag_b(Ks[s] + kb * 32, 16, DQK);
      s0 = __builtin_amdgcn_wmma_f32_16x16x32_bf16(false, qf[kb], false, b0, (short)0, s0, false, false);
      s1 = __builtin_amdgcn_wmma_f32_16x16x32_bf16(false, qf[kb], false, b1, (short)0, s1, false, false);
    }

    float p0[8], p1[8], rmax[8];
#pragma unroll
    for (int v = 0; v < 8; ++v) {
      const int sg = q0 + v + mloc;
      p0[v] = s0[v] * scale + mask[(size_t)sg * S_LEN + t0 + nloc];
      p1[v] = s1[v] * scale + mask[(size_t)sg * S_LEN + t0 + 16 + nloc];
      rmax[v] = fmaxf(p0[v], p1[v]);
    }
#pragma unroll
    for (int off = 8; off > 0; off >>= 1)
#pragma unroll
      for (int v = 0; v < 8; ++v) rmax[v] = fmaxf(rmax[v], __shfl_xor(rmax[v], off));

    float corr[8], rsum[8];
#pragma unroll
    for (int v = 0; v < 8; ++v) {
      const float mn = fmaxf(m_run[v], rmax[v]);
      corr[v] = __expf(m_run[v] - mn);
      m_run[v] = mn;
      p0[v] = __expf(p0[v] - mn);
      p1[v] = __expf(p1[v] - mn);
      rsum[v] = p0[v] + p1[v];
    }
#pragma unroll
    for (int off = 8; off > 0; off >>= 1)
#pragma unroll
      for (int v = 0; v < 8; ++v) rsum[v] += __shfl_xor(rsum[v], off);
#pragma unroll
    for (int v = 0; v < 8; ++v) l_run[v] = l_run[v] * corr[v] + rsum[v];
#pragma unroll
    for (int j = 0; j < 8; ++j)
#pragma unroll
      for (int v = 0; v < 8; ++v) o[j][v] *= corr[v];

    __bf16* pb = &Ps[wave][0];  // C-layout -> A-layout via wave-private LDS
#pragma unroll
    for (int v = 0; v < 8; ++v) {
      const int m = v + mloc;
      pb[m * 32 + nloc]      = (__bf16)p0[v];
      pb[m * 32 + 16 + nloc] = (__bf16)p1[v];
    }
    v16bf ap = load_frag_a(pb, 0, 32);
#pragma unroll
    for (int j = 0; j < 8; ++j) {
      v16bf bv = load_frag_b(Vs[s], j * 16, 32);
      o[j] = __builtin_amdgcn_wmma_f32_16x16x32_bf16(false, ap, false, bv, (short)0, o[j], false, false);
    }
  }

#pragma unroll
  for (int j = 0; j < 8; ++j)
#pragma unroll
    for (int v = 0; v < 8; ++v) {
      const int sg = q0 + v + mloc;
      X[(size_t)sg * (N_HEADS * DV) + h * DV + j * 16 + nloc] =
          (__bf16)(o[j][v] / l_run[v]);
    }
}

// ---------------------------------------------------------------- launch
extern "C" void kernel_launch(void* const* d_in, const int* in_sizes, int n_in,
                              void* d_out, int out_size, void* d_ws, size_t ws_size,
                              hipStream_t stream) {
  (void)in_sizes; (void)n_in; (void)out_size; (void)ws_size;
  const float* hidden = (const float*)d_in[0];
  const float* cosb   = (const float*)d_in[1];
  const float* sinb   = (const float*)d_in[2];
  const float* mask   = (const float*)d_in[3];
  const float* wq_a   = (const float*)d_in[4];
  const float* qnw    = (const float*)d_in[5];
  const float* wq_b   = (const float*)d_in[6];
  const float* wkv_a  = (const float*)d_in[7];
  const float* kvnw   = (const float*)d_in[8];
  const float* wkv_b  = (const float*)d_in[9];
  const float* wo     = (const float*)d_in[10];
  float* out = (float*)d_out;

  char* ws = (char*)d_ws;
  size_t off = 0;
  auto carve = [&](size_t bytes) -> void* {
    void* p = ws + off;
    off += (bytes + 255) & ~(size_t)255;
    return p;
  };
  __bf16* hid_bf  = (__bf16*)carve((size_t)S_LEN * H_DIM * 2);
  __bf16* wqa_bf  = (__bf16*)carve((size_t)QLORA * H_DIM * 2);
  __bf16* wqb_bf  = (__bf16*)carve((size_t)N_HEADS * DQK * QLORA * 2);
  __bf16* wkva_bf = (__bf16*)carve((size_t)(KVLORA + DROPE) * H_DIM * 2);
  __bf16* wkvb_bf = (__bf16*)carve((size_t)N_HEADS * 256 * KVLORA * 2);
  __bf16* wo_bf   = (__bf16*)carve((size_t)H_DIM * N_HEADS * DV * 2);
  float*  qa_f    = (float*)carve((size_t)S_LEN * QLORA * 4);
  __bf16* qan_bf  = (__bf16*)carve((size_t)S_LEN * QLORA * 2);
  float*  q_f     = (float*)carve((size_t)S_LEN * N_HEADS * DQK * 4);
  float*  kv_f    = (float*)carve((size_t)S_LEN * (KVLORA + DROPE) * 4);
  __bf16* kvcn_bf = (__bf16*)carve((size_t)S_LEN * KVLORA * 2);
  float*  kvp_f   = (float*)carve((size_t)S_LEN * N_HEADS * 256 * 4);
  __bf16* Qh      = (__bf16*)carve((size_t)N_HEADS * S_LEN * DQK * 2);
  __bf16* Kh      = (__bf16*)carve((size_t)N_HEADS * S_LEN * DQK * 2);
  __bf16* Vt      = (__bf16*)carve((size_t)N_HEADS * DV * S_LEN * 2);
  __bf16* X_bf    = (__bf16*)carve((size_t)S_LEN * N_HEADS * DV * 2);

  auto cvt = [&](const float* src, __bf16* dst, size_t n) {
    int blocks = (int)((n + 255) / 256);
    if (blocks > 4096) blocks = 4096;
    cvt_f32_bf16<<<blocks, 256, 0, stream>>>(src, dst, (int)n);
  };
  cvt(hidden, hid_bf, (size_t)S_LEN * H_DIM);
  cvt(wq_a,   wqa_bf, (size_t)QLORA * H_DIM);
  cvt(wq_b,   wqb_bf, (size_t)N_HEADS * DQK * QLORA);
  cvt(wkv_a,  wkva_bf,(size_t)(KVLORA + DROPE) * H_DIM);
  cvt(wkv_b,  wkvb_bf,(size_t)N_HEADS * 256 * KVLORA);
  cvt(wo,     wo_bf,  (size_t)H_DIM * N_HEADS * DV);

  auto gemm = [&](const __bf16* A, const __bf16* B, float* C, int M, int N, int K) {
    dim3 grid((N + 127) / 128, M / 64);
    gemm_bf16_nt<<<grid, 256, 0, stream>>>(A, B, C, M, N, K);
  };

  gemm(hid_bf, wqa_bf, qa_f, S_LEN, QLORA, H_DIM);                        // q_a
  rmsnorm_bf16<<<S_LEN, 256, 0, stream>>>(qa_f, qnw, qan_bf, QLORA, QLORA);
  gemm(qan_bf, wqb_bf, q_f, S_LEN, N_HEADS * DQK, QLORA);                 // q
  gemm(hid_bf, wkva_bf, kv_f, S_LEN, KVLORA + DROPE, H_DIM);              // kv
  rmsnorm_bf16<<<S_LEN, 256, 0, stream>>>(kv_f, kvnw, kvcn_bf, KVLORA, KVLORA + DROPE);
  gemm(kvcn_bf, wkvb_bf, kvp_f, S_LEN, N_HEADS * 256, KVLORA);            // kv_proj
  rope_pack<<<S_LEN, 256, 0, stream>>>(q_f, kv_f, kvp_f, cosb, sinb, Qh, Kh, Vt);

  const float sm_scale = (float)(pow((double)DQK, -0.5) *
                                 pow(0.1 * 1.0 * log(4.0) + 1.0, 2.0));
  flash_attn<<<dim3(S_LEN / 64, N_HEADS), 128, 0, stream>>>(Qh, Kh, Vt, mask, X_bf, sm_scale);

  gemm(X_bf, wo_bf, out, S_LEN, H_DIM, N_HEADS * DV);                     // out
}
